// SeparableConv3d_472446403146
// MI455X (gfx1250) — matrix operand order repaired and checked by reading, not compiled, not run on hardware
//
#include <hip/hip_runtime.h>
#include <hip/hip_bf16.h>

#define B_    8
#define N_    8192
#define K_    32
#define NBINS 27
#define C_    64
#define DM_   2
#define OUT_  128
#define KDIM  128            // C*DM
#define P_    (B_ * N_)      // 65536 points
#define BN_EPS 1e-5f

typedef __attribute__((ext_vector_type(16))) __bf16 v16bf;
typedef __attribute__((ext_vector_type(8)))  float  v8f;

// --- f32 -> bf16 split helpers (truncation split: x ~= hi + lo at ~f32 precision)
__device__ __forceinline__ float bf16_trunc_f(float x) {
    union { float f; unsigned u; } a; a.f = x; a.u &= 0xFFFF0000u; return a.f;
}
__device__ __forceinline__ unsigned bf16_pack2(float x0, float x1) {
    union { float f; unsigned u; } a0, a1; a0.f = x0; a1.f = x1;
    return (a0.u >> 16) | (a1.u & 0xFFFF0000u);   // e_even in [15:0], e_odd in [31:16]
}

// =====================================================================
// Kernel 0: one-time pack of fc_w into WMMA B-fragment layout, pre-split
// into bf16 hi/lo planes.  Layout (dwords): [wcol 8][kt 4][lane 32][j 8],
// hi plane first (8192 dwords), lo plane next (8192 dwords).
// B-frag element e (0..15) of lane L maps to K = kt*32 + 16*(L>>4) + e,
// O = wcol*16 + (L&15); dword j packs elements 2j (lo16) and 2j+1 (hi16).
// =====================================================================
__global__ __launch_bounds__(256) void pack_fcw(
    const float* __restrict__ fc_w, unsigned* __restrict__ bw)
{
    const int t    = blockIdx.x * 256 + threadIdx.x;  // 0..8191
    const int j    = t & 7;
    const int lane = (t >> 3) & 31;
    const int kt   = (t >> 8) & 3;
    const int wcol = t >> 10;
    const int o    = wcol * 16 + (lane & 15);
    const int k    = kt * 32 + 16 * (lane >> 4) + 2 * j;

    const float x0 = fc_w[k * OUT_ + o];
    const float x1 = fc_w[(k + 1) * OUT_ + o];
    const float h0 = bf16_trunc_f(x0);
    const float h1 = bf16_trunc_f(x1);

    bw[t]        = bf16_pack2(x0, x1);            // hi plane
    bw[8192 + t] = bf16_pack2(x0 - h0, x1 - h1);  // lo plane
}

// =====================================================================
// Kernel 1 (fused): gather/bin/depthwise (VALU+LDS) -> FC (WMMA bf16x3)
// Block = 256 threads = 8 waves, handles 16 consecutive points.
//   Phase 1: wave w bins points 2w, 2w+1 into private LDS slice, applies
//            the depthwise kernel, and writes conv rows to LDS already
//            SPLIT into packed bf16 hi/lo pair planes (split done once,
//            not once per consuming wave).
//   Phase 2: wave w computes outputs [16w,16w+16) for all 16 points via
//            3x v_wmma_f32_16x16x32_bf16 per K-tile.  A fragments are
//            2x ds_load_b128 per plane, B fragments 2x global b128 from
//            the pre-packed weights -- no conversion VALU in the loop.
// Pre-BN result y goes to d_out (overwritten later by BN+ReLU pass).
// =====================================================================
__global__ __launch_bounds__(256) void fused_conv_fc(
    const float* __restrict__ inputs, const int* __restrict__ nn_index,
    const int* __restrict__ nn_count, const int* __restrict__ filt_index,
    const float* __restrict__ dk, const unsigned* __restrict__ bw,
    const float* __restrict__ fc_b, float* __restrict__ y)
{
    __shared__ __align__(16) float    bins_lds[8 * NBINS * C_]; // 55296 B
    __shared__ __align__(16) unsigned convHi[16 * 64];          // 4096 B (bf16 pairs)
    __shared__ __align__(16) unsigned convLo[16 * 64];          // 4096 B

    const int tid   = threadIdx.x;
    const int lane  = tid & 31;
    const int wv    = tid >> 5;
    const int tileP = blockIdx.x * 16;
    float* bins = bins_lds + wv * (NBINS * C_);

    // ---------------- Phase 1: binning + depthwise + split ----------------
    const int c0 = lane, c1 = lane + 32;   // each lane owns 2 channels
    for (int s = 0; s < 2; ++s) {
        const int lp = 2 * wv + s;
        const int p  = tileP + lp;

        for (int i = lane; i < NBINS * C_; i += 32) bins[i] = 0.0f;

        const int cnt     = nn_count[p];            // uniform across wave, in [1,K]
        const int rowBase = (p >> 13) << 13;        // b * N  (N = 8192)

        for (int k = 0; k < cnt; ++k) {
            const int idx = nn_index[p * K_ + k];
            const int fi  = filt_index[p * K_ + k]; // uniform -> conflict-free banks
            const float* src = inputs + (size_t)(rowBase + idx) * C_;
            bins[fi * C_ + c0] += src[c0];
            bins[fi * C_ + c1] += src[c1];
        }

        const float rcp = 1.0f / (float)cnt;
        float a00 = 0.f, a01 = 0.f, a10 = 0.f, a11 = 0.f;
        #pragma unroll
        for (int sb = 0; sb < NBINS; ++sb) {
            const float  v0 = bins[sb * C_ + c0];
            const float  v1 = bins[sb * C_ + c1];
            const float2 w0 = ((const float2*)dk)[sb * C_ + c0]; // dk[sb][c][0..1]
            const float2 w1 = ((const float2*)dk)[sb * C_ + c1];
            a00 += v0 * w0.x; a01 += v0 * w0.y;
            a10 += v1 * w1.x; a11 += v1 * w1.y;
        }
        // conv[lp][2c0], [2c0+1] -> pair dword `c0`; conv[lp][2c1..] -> pair `c1`
        const float x00 = a00 * rcp, x01 = a01 * rcp;
        const float x10 = a10 * rcp, x11 = a11 * rcp;
        const float h00 = bf16_trunc_f(x00), h01 = bf16_trunc_f(x01);
        const float h10 = bf16_trunc_f(x10), h11 = bf16_trunc_f(x11);
        convHi[lp * 64 + c0] = bf16_pack2(x00, x01);
        convLo[lp * 64 + c0] = bf16_pack2(x00 - h00, x01 - h01);
        convHi[lp * 64 + c1] = bf16_pack2(x10, x11);
        convLo[lp * 64 + c1] = bf16_pack2(x10 - h10, x11 - h11);
    }
    __syncthreads();

    // ---------------- Phase 2: FC via WMMA (bf16 hi/lo split) ----------------
    const int colBase = wv * 16;
    const int mrow    = lane & 15;   // A row and B/D column within tile (both lane halves)
    const int g       = lane >> 4;   // lane-half selector
    v8f acc = {};

    #pragma unroll
    for (int kt = 0; kt < 4; ++kt) {
        // A fragment (16x32 bf16, contiguous K = k0+16g+e, same as B), as packed K-pairs:
        //  frag dwords 0..3 = pairs (k0+16g)/2 .. +3   (elements 0..7)
        //  frag dwords 4..7 = pairs (k0+16g+8)/2 .. +3 (elements 8..15)
        const int pairBase = mrow * 64 + kt * 16 + 8 * g;
        union { uint4 q[2]; v16bf v; } AH, AL;
        AH.q[0] = *(const uint4*)(convHi + pairBase);
        AH.q[1] = *(const uint4*)(convHi + pairBase + 4);
        AL.q[0] = *(const uint4*)(convLo + pairBase);
        AL.q[1] = *(const uint4*)(convLo + pairBase + 4);

        // B fragments: pre-packed, lane-contiguous 32B -> two b128 loads each
        const unsigned fragOff = ((unsigned)(wv * 4 + kt) * 32 + (unsigned)lane) * 8u;
        union { uint4 q[2]; v16bf v; } BH, BL;
        BH.q[0] = *(const uint4*)(bw + fragOff);
        BH.q[1] = *(const uint4*)(bw + fragOff + 4);
        BL.q[0] = *(const uint4*)(bw + 8192 + fragOff);
        BL.q[1] = *(const uint4*)(bw + 8192 + fragOff + 4);

        // D = A*B + C at ~f32 precision: Ahi*Bhi + Ahi*Blo + Alo*Bhi
        acc = __builtin_amdgcn_wmma_f32_16x16x32_bf16(false, AH.v, false, BH.v, (short)0, acc, false, false);
        acc = __builtin_amdgcn_wmma_f32_16x16x32_bf16(false, AH.v, false, BL.v, (short)0, acc, false, false);
        acc = __builtin_amdgcn_wmma_f32_16x16x32_bf16(false, AL.v, false, BH.v, (short)0, acc, false, false);
    }

    // C/D layout: VGPR r -> rows r (lanes 0-15) and r+8 (lanes 16-31)
    const int   col  = colBase + mrow;
    const float bias = fc_b[col];
    #pragma unroll
    for (int r = 0; r < 8; ++r) {
        const int prow = tileP + r + 8 * g;
        y[(size_t)prow * OUT_ + col] = acc[r] + bias;
    }
}

// =====================================================================
// Kernel 2: per-channel BN stats (deterministic: fixed-order partials +
// LDS tree; no float atomics). One block per output channel.
// =====================================================================
__global__ __launch_bounds__(256) void bn_stats(
    const float* __restrict__ y, const float* __restrict__ gamma,
    const float* __restrict__ beta, float* __restrict__ scale,
    float* __restrict__ shift)
{
    __shared__ float s1[256], s2[256];
    const int ch = blockIdx.x, t = threadIdx.x;
    float a = 0.0f, b = 0.0f;
    for (int p = t; p < P_; p += 256) {
        const float v = y[(size_t)p * OUT_ + ch];
        a += v; b += v * v;
    }
    s1[t] = a; s2[t] = b;
    __syncthreads();
    for (int st = 128; st > 0; st >>= 1) {
        if (t < st) { s1[t] += s1[t + st]; s2[t] += s2[t + st]; }
        __syncthreads();
    }
    if (t == 0) {
        const float inv  = 1.0f / (float)P_;
        const float mean = s1[0] * inv;
        const float var  = s2[0] * inv - mean * mean;
        const float sc   = gamma[ch] * rsqrtf(var + BN_EPS);
        scale[ch] = sc;
        shift[ch] = beta[ch] - mean * sc;
    }
}

// =====================================================================
// Kernel 3: in-place BN apply + ReLU over d_out.
// =====================================================================
__global__ __launch_bounds__(256) void bn_apply(
    float* __restrict__ y, const float* __restrict__ scale,
    const float* __restrict__ shift)
{
    const int i  = blockIdx.x * 256 + threadIdx.x;
    const int ch = i & (OUT_ - 1);
    const float v = y[i] * scale[ch] + shift[ch];
    y[i] = fmaxf(v, 0.0f);
}

extern "C" void kernel_launch(void* const* d_in, const int* in_sizes, int n_in,
                              void* d_out, int out_size, void* d_ws, size_t ws_size,
                              hipStream_t stream) {
    (void)in_sizes; (void)n_in; (void)out_size; (void)ws_size;
    const float* inputs     = (const float*)d_in[0];
    const int*   nn_index   = (const int*)d_in[1];
    const int*   nn_count   = (const int*)d_in[2];
    const int*   filt_index = (const int*)d_in[3];
    const float* dk         = (const float*)d_in[4];
    const float* fc_w       = (const float*)d_in[5];
    const float* fc_b       = (const float*)d_in[6];
    const float* gamma      = (const float*)d_in[7];
    const float* beta       = (const float*)d_in[8];

    float*    y     = (float*)d_out;            // pre-BN y lives in d_out, finished in place
    float*    scale = (float*)d_ws;             // 128 floats
    float*    shift = scale + OUT_;             // 128 floats
    unsigned* bw    = (unsigned*)(scale + 256); // packed bf16 hi/lo fc_w: 16384 dwords (64 KB)

    pack_fcw<<<32, 256, 0, stream>>>(fc_w, bw);
    fused_conv_fc<<<P_ / 16, 256, 0, stream>>>(inputs, nn_index, nn_count,
                                               filt_index, dk, bw, fc_b, y);
    bn_stats<<<OUT_, 256, 0, stream>>>(y, gamma, beta, scale, shift);
    bn_apply<<<(P_ * OUT_) / 256, 256, 0, stream>>>(y, scale, shift);
}
